// CrossAttention_2594160247068
// MI455X (gfx1250) — compile-verified, matrix operand order checked by reading, not run on hardware
//
#include <hip/hip_runtime.h>
#include <hip/hip_bf16.h>

// ---------------------------------------------------------------------------
// CDNA5 (gfx1250, wave32) fused cross-attention:
//   LN -> Q/KV projections (f16 WMMA, double-buffered) ->
//   flash attention (f16 WMMA, online softmax, null-token init,
//                    async-LDS V staging overlapped with S/softmax) ->
//   out projection (f16 WMMA, f32 out) -> LN
// ---------------------------------------------------------------------------

typedef __attribute__((ext_vector_type(16))) _Float16 v16h;
typedef __attribute__((ext_vector_type(8)))  _Float16 v8h;
typedef __attribute__((ext_vector_type(8)))  float    v8f;

#define BATCH   4
#define NQ      2048
#define MCTX    2048
#define DIM     1024
#define HEADS   8
#define DH      64
#define INNER   512            // HEADS*DH
#define BN      (BATCH * NQ)   // 8192 flattened rows
#define SCALE   0.125f         // 64^-0.5

// ---- f16 WMMA fragment helpers --------------------------------------------
// 16-bit A/B fragment (16x32 / 32x16): per lane (half = lane>>4) the 16 f16
// values are the contiguous chunks [half*8 .. half*8+7] and [16+half*8 .. +7]
// relative to the fragment's K base.  `p` must already point at
// base + k0 + half*8 (16B aligned).
__device__ __forceinline__ v16h load_frag(const _Float16* p) {
    v8h lo = *(const v8h*)(p);
    v8h hi = *(const v8h*)(p + 16);
    v16h r;
#pragma unroll
    for (int i = 0; i < 8; ++i) { r[i] = lo[i]; r[i + 8] = hi[i]; }
    return r;
}

__device__ __forceinline__ v8f wmma_f16(v16h a, v16h b, v8f c) {
    return __builtin_amdgcn_wmma_f32_16x16x32_f16(
        /*neg_a=*/false, a, /*neg_b=*/false, b,
        /*c_mod=*/(short)0, c, /*reuse_a=*/false, /*reuse_b=*/false);
}

// Row-wise reductions over the 16-lane half-groups of the C/D layout.
// xor masks 1,2,4,8 stay inside lanes [0..15] / [16..31] on wave32.
__device__ __forceinline__ float rowmax16(float x) {
#pragma unroll
    for (int m = 1; m < 16; m <<= 1) x = fmaxf(x, __shfl_xor(x, m, 32));
    return x;
}
__device__ __forceinline__ float rowsum16(float x) {
#pragma unroll
    for (int m = 1; m < 16; m <<= 1) x += __shfl_xor(x, m, 32);
    return x;
}

// ---- elementwise / prep kernels -------------------------------------------
__global__ __launch_bounds__(256) void cast_f32_to_f16_kernel(
    const float* __restrict__ src, _Float16* __restrict__ dst, int n) {
    int i = blockIdx.x * 256 + threadIdx.x;
    if (i < n) dst[i] = (_Float16)src[i];
}

// src is [R,C] f32 row-major; dst is [C,R] f16 (i.e. src transposed) so that
// WMMA B-fragments become contiguous 16-byte loads along K.
__global__ __launch_bounds__(256) void transpose_cast_kernel(
    const float* __restrict__ src, _Float16* __restrict__ dst, int R, int C) {
    int i = blockIdx.x * 256 + threadIdx.x;
    if (i >= R * C) return;
    int r = i / C, c = i - r * C;
    dst[(size_t)c * R + r] = (_Float16)src[i];
}

template <typename OUT>
__global__ __launch_bounds__(256) void layernorm_kernel(
    const float* __restrict__ x, const float* __restrict__ gamma,
    OUT* __restrict__ out, int D) {
    __shared__ float red1[256];
    __shared__ float red2[256];
    int row = blockIdx.x;
    const float* xr = x + (size_t)row * D;
    float s = 0.f, s2 = 0.f;
    for (int i = threadIdx.x; i < D; i += 256) {
        float v = xr[i]; s += v; s2 += v * v;
    }
    red1[threadIdx.x] = s; red2[threadIdx.x] = s2;
    __syncthreads();
#pragma unroll
    for (int off = 128; off > 0; off >>= 1) {
        if (threadIdx.x < off) {
            red1[threadIdx.x] += red1[threadIdx.x + off];
            red2[threadIdx.x] += red2[threadIdx.x + off];
        }
        __syncthreads();
    }
    float mu  = red1[0] * (1.0f / D);
    float var = red2[0] * (1.0f / D) - mu * mu;
    float inv = rsqrtf(var + 1e-5f);
    for (int i = threadIdx.x; i < D; i += 256)
        out[(size_t)row * D + i] = (OUT)((xr[i] - mu) * inv * gamma[i]);
}

// ---- WMMA GEMM: C[M,N] = A[M,K] (f16) @ Bt[N,K]^T (f16) -------------------
// One wave computes a 16(M) x 64(N) macro-tile.  K in steps of 32 with
// register double-buffering: fragments for step k+1 are issued before the
// WMMAs of step k so s_wait_loadcnt overlaps the matrix pipe.
template <bool F32OUT>
__global__ __launch_bounds__(32) void wmma_gemm_kernel(
    const _Float16* __restrict__ A, const _Float16* __restrict__ Bt,
    void* __restrict__ C, int M, int N, int K) {
    int lane = threadIdx.x;
    int lm = lane & 15, half = lane >> 4;
    int tile_m = blockIdx.y * 16;
    int tile_n = blockIdx.x * 64;

    v8f acc[4] = {v8f{}, v8f{}, v8f{}, v8f{}};
    const _Float16* arow = A + (size_t)(tile_m + lm) * K + half * 8;
    const _Float16* brow[4];
#pragma unroll
    for (int t = 0; t < 4; ++t)
        brow[t] = Bt + (size_t)(tile_n + t * 16 + lm) * K + half * 8;

    v16h a_cur = load_frag(arow);
    v16h b_cur[4];
#pragma unroll
    for (int t = 0; t < 4; ++t) b_cur[t] = load_frag(brow[t]);

    for (int k0 = 0; k0 < K; k0 += 32) {
        int kn = k0 + 32;
        v16h a_nxt = a_cur;
        v16h b_nxt[4] = {b_cur[0], b_cur[1], b_cur[2], b_cur[3]};
        if (kn < K) {                      // issue next-step loads first
            a_nxt = load_frag(arow + kn);
#pragma unroll
            for (int t = 0; t < 4; ++t) b_nxt[t] = load_frag(brow[t] + kn);
        }
#pragma unroll
        for (int t = 0; t < 4; ++t) acc[t] = wmma_f16(a_cur, b_cur[t], acc[t]);
        a_cur = a_nxt;
#pragma unroll
        for (int t = 0; t < 4; ++t) b_cur[t] = b_nxt[t];
    }
    // C/D layout: lane holds column tile_n + t*16 + lm, VGPR v -> row v+8*half
#pragma unroll
    for (int t = 0; t < 4; ++t)
#pragma unroll
        for (int v = 0; v < 8; ++v) {
            size_t idx = (size_t)(tile_m + v + 8 * half) * N + tile_n + t * 16 + lm;
            if (F32OUT) ((float*)C)[idx]     = acc[t][v];
            else        ((_Float16*)C)[idx]  = (_Float16)acc[t][v];
        }
}

// ---- flash attention with null token --------------------------------------
// grid = (NQ/16, HEADS, BATCH), 1 wave per block.
// q  : [BN, INNER] f16     kv : [BN, 2*INNER] f16 (k | v)
// nullkv : [2,64] f32      mask : [BATCH, MCTX] (1 byte each, jax bool)
// out: [BN, INNER] f16
__global__ __launch_bounds__(32) void attn_kernel(
    const _Float16* __restrict__ q, const _Float16* __restrict__ kv,
    const float* __restrict__ nullkv, const unsigned char* __restrict__ mask,
    _Float16* __restrict__ out) {
    const int DKV = 2 * INNER;
    int lane = threadIdx.x;
    int lm = lane & 15, half = lane >> 4;
    int qt = blockIdx.x, hh = blockIdx.y, bb = blockIdx.z;
    size_t qbase = (size_t)bb * NQ + qt * 16;
    size_t kbase = (size_t)bb * NQ;

    __shared__ __align__(16) float    s_ld[16];
    __shared__ __align__(16) _Float16 p_ld[16 * 32];   // P tile transpose bounce
    __shared__ __align__(16) _Float16 v_ld[32 * 64];   // V tile (key-major)

    // Per-lane LDS destination for the async V-tile DMA: for LDS-aperture flat
    // addresses, addr[31:0] is the wave-relative LDS byte offset (ISA 10.2).
    unsigned v_lds_dst = (unsigned)(uintptr_t)(&v_ld[lane * DH]);

    // Q A-fragments: row lm of the 16-query tile, 64 features in two K=32 frags
    const _Float16* qp = q + (qbase + lm) * INNER + hh * DH;
    v16h aq0 = load_frag(qp + half * 8);
    v16h aq1 = load_frag(qp + 32 + half * 8);

    // null token -> softmax initial state: m = scale*q.null_k, l = 1, O = null_v
    const float* nk = nullkv;
    const float* nv = nullkv + DH;
    if (lane < 16) {
        const _Float16* qr = q + (qbase + lane) * INNER + hh * DH;
        float s = 0.f;
#pragma unroll
        for (int d = 0; d < DH; ++d) s += (float)qr[d] * nk[d];
        s_ld[lane] = s * SCALE;
    }
    __syncthreads();

    float m_row[8], l_row[8];
#pragma unroll
    for (int v = 0; v < 8; ++v) { m_row[v] = s_ld[v + 8 * half]; l_row[v] = 1.0f; }
    v8f acc[4];
#pragma unroll
    for (int t = 0; t < 4; ++t) {
        float nvv = nv[t * 16 + lm];
#pragma unroll
        for (int v = 0; v < 8; ++v) acc[t][v] = nvv;
    }

    const unsigned char* mrow = mask + (size_t)bb * MCTX;

    for (int j0 = 0; j0 < MCTX; j0 += 32) {
        __syncthreads();   // previous iteration done reading p_ld / v_ld

        // ---- kick off async DMA of the V tile into LDS (ASYNCcnt path). ----
        // Lane copies its 128-byte key row (key j0+lane); INST_OFFSET advances
        // both the global and LDS addresses, so one base pair + 8 offsets.
        {
            const _Float16* vp = kv + (kbase + j0 + lane) * DKV + INNER + hh * DH;
            unsigned long long ga = (unsigned long long)(uintptr_t)vp;
            asm volatile(
                "global_load_async_to_lds_b128 %0, %1, off\n\t"
                "global_load_async_to_lds_b128 %0, %1, off offset:16\n\t"
                "global_load_async_to_lds_b128 %0, %1, off offset:32\n\t"
                "global_load_async_to_lds_b128 %0, %1, off offset:48\n\t"
                "global_load_async_to_lds_b128 %0, %1, off offset:64\n\t"
                "global_load_async_to_lds_b128 %0, %1, off offset:80\n\t"
                "global_load_async_to_lds_b128 %0, %1, off offset:96\n\t"
                "global_load_async_to_lds_b128 %0, %1, off offset:112"
                :
                : "v"(v_lds_dst), "v"(ga)
                : "memory");
        }
        if (j0 + 32 < MCTX)  // prefetch next K tile rows
            __builtin_prefetch(kv + (kbase + j0 + 32 + lane) * DKV + hh * DH, 0, 0);

        // ---- S = scale * Q @ K^T (two 16-key C tiles); overlaps V DMA ------
        v16h bk[2][2];
#pragma unroll
        for (int u = 0; u < 2; ++u) {
            const _Float16* kp = kv + (kbase + j0 + u * 16 + lm) * DKV + hh * DH;
            bk[u][0] = load_frag(kp + half * 8);
            bk[u][1] = load_frag(kp + 32 + half * 8);
        }
        v8f s[2];
#pragma unroll
        for (int u = 0; u < 2; ++u) {
            v8f z = v8f{};
            z = wmma_f16(aq0, bk[u][0], z);
            z = wmma_f16(aq1, bk[u][1], z);
            float mpen = mrow[j0 + u * 16 + lm] ? 0.f : -1e30f;  // lane == key col
#pragma unroll
            for (int v = 0; v < 8; ++v) z[v] = z[v] * SCALE + mpen;
            s[u] = z;
        }
        // online softmax update (rows live in half-groups of 16 lanes)
#pragma unroll
        for (int v = 0; v < 8; ++v) {
            float mx = rowmax16(fmaxf(s[0][v], s[1][v]));
            float mnew = fmaxf(m_row[v], mx);
            float p0 = __expf(s[0][v] - mnew);
            float p1 = __expf(s[1][v] - mnew);
            s[0][v] = p0; s[1][v] = p1;
            float rs   = rowsum16(p0 + p1);
            float corr = __expf(m_row[v] - mnew);
            l_row[v] = l_row[v] * corr + rs;
            m_row[v] = mnew;
#pragma unroll
            for (int t = 0; t < 4; ++t) acc[t][v] *= corr;
        }
        // C/D layout -> A-fragment layout via LDS bounce (store P as f16)
#pragma unroll
        for (int u = 0; u < 2; ++u)
#pragma unroll
            for (int v = 0; v < 8; ++v)
                p_ld[(v + 8 * half) * 32 + u * 16 + lm] = (_Float16)s[u][v];

        // V DMA must have landed before we gather B-fragments from v_ld.
        asm volatile("s_wait_asynccnt 0" ::: "memory");
        __syncthreads();
        v16h ap = load_frag(&p_ld[lm * 32 + half * 8]);

        // O += P @ V   (V B-fragments gathered from LDS: lane = d column)
#pragma unroll
        for (int t = 0; t < 4; ++t) {
            v16h bv;
#pragma unroll
            for (int i = 0; i < 8; ++i) {
                bv[i]     = v_ld[(half * 8 + i) * DH      + t * 16 + lm];
                bv[i + 8] = v_ld[(16 + half * 8 + i) * DH + t * 16 + lm];
            }
            acc[t] = wmma_f16(ap, bv, acc[t]);
        }
    }

    // epilogue: O / l, merge heads back into [BN, INNER]
#pragma unroll
    for (int t = 0; t < 4; ++t)
#pragma unroll
        for (int v = 0; v < 8; ++v) {
            float o = acc[t][v] / l_row[v];
            out[(qbase + v + 8 * half) * INNER + hh * DH + t * 16 + lm] = (_Float16)o;
        }
}

// ---------------------------------------------------------------------------
extern "C" void kernel_launch(void* const* d_in, const int* in_sizes, int n_in,
                              void* d_out, int out_size, void* d_ws, size_t ws_size,
                              hipStream_t stream) {
    const float*         x        = (const float*)d_in[0];
    const float*         context  = (const float*)d_in[1];
    const unsigned char* mask     = (const unsigned char*)d_in[2]; // jax bool = 1 byte
    const float*         gamma    = (const float*)d_in[3];
    const float*         w_q      = (const float*)d_in[4];
    const float*         w_kv     = (const float*)d_in[5];
    const float*         null_kv  = (const float*)d_in[6];
    const float*         w_out    = (const float*)d_in[7];
    const float*         out_g    = (const float*)d_in[8];
    float*               out      = (float*)d_out;

    char* ws = (char*)d_ws;
    size_t off = 0;
    auto alloc = [&](size_t bytes) -> void* {
        off = (off + 255) & ~(size_t)255;
        void* p = ws + off;
        off += bytes;
        return p;
    };
    _Float16* xn    = (_Float16*)alloc((size_t)BN * DIM * 2);
    _Float16* ctx16 = (_Float16*)alloc((size_t)BN * DIM * 2);
    _Float16* wqT   = (_Float16*)alloc((size_t)INNER * DIM * 2);      // [512,1024]
    _Float16* wkvT  = (_Float16*)alloc((size_t)DIM * DIM * 2);        // [1024,1024]
    _Float16* woutT = (_Float16*)alloc((size_t)DIM * INNER * 2);      // [1024,512]
    _Float16* q16   = (_Float16*)alloc((size_t)BN * INNER * 2);
    _Float16* kv16  = (_Float16*)alloc((size_t)BN * DIM * 2);
    _Float16* ao16  = (_Float16*)alloc((size_t)BN * INNER * 2);
    float*    proj  = (float*)   alloc((size_t)BN * DIM * 4);

    // 1) pre-norm of x -> f16
    layernorm_kernel<_Float16><<<BN, 256, 0, stream>>>(x, gamma, xn, DIM);
    // 2) cast context, cast+transpose weights
    {
        int n = BN * DIM;
        cast_f32_to_f16_kernel<<<(n + 255) / 256, 256, 0, stream>>>(context, ctx16, n);
        n = DIM * INNER;
        transpose_cast_kernel<<<(n + 255) / 256, 256, 0, stream>>>(w_q, wqT, DIM, INNER);
        n = DIM * DIM;
        transpose_cast_kernel<<<(n + 255) / 256, 256, 0, stream>>>(w_kv, wkvT, DIM, DIM);
        n = INNER * DIM;
        transpose_cast_kernel<<<(n + 255) / 256, 256, 0, stream>>>(w_out, woutT, INNER, DIM);
    }
    // 3) Q and KV projections (f16 WMMA, f16 out)
    wmma_gemm_kernel<false><<<dim3(INNER / 64, BN / 16), 32, 0, stream>>>(
        xn, wqT, q16, BN, INNER, DIM);
    wmma_gemm_kernel<false><<<dim3(DIM / 64, BN / 16), 32, 0, stream>>>(
        ctx16, wkvT, kv16, BN, DIM, DIM);
    // 4) fused flash attention with null token (async-LDS V staging)
    attn_kernel<<<dim3(NQ / 16, HEADS, BATCH), 32, 0, stream>>>(
        q16, kv16, null_kv, mask, ao16);
    // 5) output projection (f32 out) + final layernorm
    wmma_gemm_kernel<true><<<dim3(DIM / 64, BN / 16), 32, 0, stream>>>(
        ao16, woutT, proj, BN, DIM, INNER);
    layernorm_kernel<float><<<BN, 256, 0, stream>>>(proj, out_g, out, DIM);
}
